// GinkaBottleneck_37409165148986
// MI455X (gfx1250) — matrix operand order repaired and checked by reading, not compiled, not run on hardware
//
#include <hip/hip_runtime.h>

typedef __attribute__((ext_vector_type(16))) _Float16 v16h;
typedef __attribute__((ext_vector_type(8)))  _Float16 v8h;
typedef __attribute__((ext_vector_type(8)))  float    v8f;

#define BATCH 2048
#define TOK 16
#define DIM 512
#define FFD 1024
#define NHEAD 8
#define NLAYER 4
#define BT (BATCH*TOK)   /* 32768 rows */
#define SB_STRIDE 40     /* padded LDS row stride (halfs): 80B, 16B-aligned, bank-spread */

static __device__ __forceinline__ v16h cat8(v8h lo, v8h hi) {
  v16h r;
#pragma unroll
  for (int i = 0; i < 8; ++i) { r[i] = lo[i]; r[8 + i] = hi[i]; }
  return r;
}

// ---------- weight convert: f32 [K,N] -> f16 [N,K] (tr=1) or flat copy ----------
__global__ __launch_bounds__(256) void k_cvt(const float* __restrict__ s,
                                             _Float16* __restrict__ d,
                                             int K, int N, int tr) {
  int i = blockIdx.x * 256 + threadIdx.x;
  if (i >= K * N) return;
  if (tr) { int k = i / N, n = i % N; d[(size_t)n * K + k] = (_Float16)s[i]; }
  else     d[i] = (_Float16)s[i];
}

// ---------- x [B,512,4,4] -> token-major f16 [B*16, 512] ----------
__global__ __launch_bounds__(256) void k_pack(const float* __restrict__ x,
                                              _Float16* __restrict__ xt) {
  int i = blockIdx.x * 256 + threadIdx.x;      // B*512*16 threads
  int t = i & 15; int c = (i >> 4) & 511; int b = i >> 13;
  xt[((size_t)b * 16 + t) * 512 + c] = (_Float16)x[i];
}

// ---------- WMMA GEMM: out[M,N] = A[M,K](f16) * W[N,K]^T (f16) + bias ----------
// block = 4 waves = 128 M-rows x 64 N-cols; wave = 32x64 (8 f32 accumulators).
// Weight panel (64n x 32k) double-buffered in LDS, shared by all 4 waves.
__global__ __launch_bounds__(128) void k_gemm(const _Float16* __restrict__ A,
                                              const _Float16* __restrict__ W,
                                              const float* __restrict__ bias,
                                              float* __restrict__ out32,
                                              _Float16* __restrict__ out16,
                                              int M, int K, int N, int act) {
  __shared__ _Float16 sB[2][64 * SB_STRIDE];
  int wv   = threadIdx.x >> 5;                 // wave in block: M-subblock
  int lane = threadIdx.x & 31;
  int g = lane >> 4, mr = lane & 15;
  int nt = N >> 6;
  int mblk = blockIdx.x / nt;
  int ng   = blockIdx.x % nt;
  int n0   = ng << 6;
  size_t mbase = (size_t)mblk * 128 + wv * 32; // this wave's 32 rows
  const _Float16* arow0 = A + (mbase + mr) * K;
  const _Float16* arow1 = A + (mbase + 16 + mr) * K;
  // staging: thread loads 32B of the weight panel: row n = tid>>1, half = (tid&1)*16
  int srow = threadIdx.x >> 1, shalf = (threadIdx.x & 1) * 16;
  const _Float16* wrow = W + (size_t)(n0 + srow) * K + shalf;
  _Float16* sdst0 = &sB[0][srow * SB_STRIDE + shalf];
  _Float16* sdst1 = &sB[1][srow * SB_STRIDE + shalf];

  v8f acc[8] = {};
  { // preload k-step 0 into buffer 0
    v8h w0 = *(const v8h*)(wrow);
    v8h w1 = *(const v8h*)(wrow + 8);
    ((v8h*)sdst0)[0] = w0;
    *(v8h*)(sdst0 + 8) = w1;
  }
  __syncthreads();

  int nsteps = K >> 5;
  for (int ks = 0; ks < nsteps; ++ks) {
    int kb = ks << 5;
    int cur = ks & 1;
    _Float16* nb = cur ? sdst0 : sdst1;
    bool stage = (ks + 1) < nsteps;
    v8h w0, w1;
    if (stage) {           // issue next panel's global loads early...
      w0 = *(const v8h*)(wrow + kb + 32);
      w1 = *(const v8h*)(wrow + kb + 40);
    }
    __builtin_prefetch(arow0 + kb + 64, 0, 3);   // global_prefetch_b8, A stream
    __builtin_prefetch(arow1 + kb + 64, 0, 3);
    // A fragments (16x32 layout: half g holds K=g*8+0..7 then 16+g*8+0..7)
    v16h a0 = cat8(*(const v8h*)(arow0 + kb + g * 8),
                   *(const v8h*)(arow0 + kb + 16 + g * 8));
    v16h a1 = cat8(*(const v8h*)(arow1 + kb + g * 8),
                   *(const v8h*)(arow1 + kb + 16 + g * 8));
#pragma unroll
    for (int t = 0; t < 4; ++t) {
      const _Float16* bp = &sB[cur][(t * 16 + mr) * SB_STRIDE + g * 8];
      v16h bm = cat8(((const v8h*)bp)[0], *(const v8h*)(bp + 16));
      acc[t]     = __builtin_amdgcn_wmma_f32_16x16x32_f16(false, a0, false, bm,
                                                          (short)0, acc[t], false, false);
      acc[4 + t] = __builtin_amdgcn_wmma_f32_16x16x32_f16(false, a1, false, bm,
                                                          (short)0, acc[4 + t], false, false);
    }
    if (stage) {           // ...consume them after the matrix math
      ((v8h*)nb)[0] = w0;
      *(v8h*)(nb + 8) = w1;
    }
    __syncthreads();
  }

#pragma unroll
  for (int s = 0; s < 2; ++s)
#pragma unroll
    for (int t = 0; t < 4; ++t) {
      int n = n0 + t * 16 + mr;
      float bv = bias ? bias[n] : 0.f;
#pragma unroll
      for (int r = 0; r < 8; ++r) {            // C layout: row = r + 8*g, col = mr
        size_t row = mbase + s * 16 + r + 8 * g;
        float v = acc[s * 4 + t][r] + bv;
        if (act == 1) v = fmaxf(v, 0.f);
        if (out32) out32[row * N + n] = v;
        if (out16) out16[row * N + n] = (_Float16)v;
      }
    }
}

// ---------- attention: one wave per (batch, head); T=16, HD=64, all WMMA ----------
__global__ __launch_bounds__(256) void k_attn(const _Float16* __restrict__ Q,
                                              const _Float16* __restrict__ Kk,
                                              const _Float16* __restrict__ V,
                                              _Float16* __restrict__ O) {
  __shared__ _Float16 sP[8][16 * 16];
  __shared__ _Float16 sV[8][16 * 64];
  int slot = threadIdx.x >> 5;
  int pair = blockIdx.x * 8 + slot;        // b*NHEAD + head
  int b = pair >> 3, head = pair & 7;
  int lane = threadIdx.x & 31, g = lane >> 4, mr = lane & 15;
  size_t base = ((size_t)b * 16) * 512 + head * 64;

  // S = Q K^T  (16x16, K-dim 64 -> two chained wmma)
  v8f s = {};
#pragma unroll
  for (int kb = 0; kb < 64; kb += 32) {
    const _Float16* qr = Q  + base + (size_t)mr * 512;
    const _Float16* kr = Kk + base + (size_t)mr * 512;  // B col n = key token = mr
    v16h a  = cat8(*(const v8h*)(qr + kb + g * 8),
                   *(const v8h*)(qr + kb + 16 + g * 8));
    v16h bm = cat8(*(const v8h*)(kr + kb + g * 8),
                   *(const v8h*)(kr + kb + 16 + g * 8));
    s = __builtin_amdgcn_wmma_f32_16x16x32_f16(false, a, false, bm, (short)0, s, false, false);
  }
  // softmax over keys: columns live across the 16-lane half (xor<=8 stays in half)
#pragma unroll
  for (int r = 0; r < 8; ++r) {
    float v = s[r] * 0.125f;               // 1/sqrt(64)
    float mx = v;
#pragma unroll
    for (int m = 8; m >= 1; m >>= 1) mx = fmaxf(mx, __shfl_xor(mx, m, 32));
    float e = __expf(v - mx);
    float sum = e;
#pragma unroll
    for (int m = 8; m >= 1; m >>= 1) sum += __shfl_xor(sum, m, 32);
    s[r] = e / sum;
  }
  // stage P (C-layout -> LDS) and V rows
#pragma unroll
  for (int r = 0; r < 8; ++r) sP[slot][(r + 8 * g) * 16 + mr] = (_Float16)s[r];
  {
    int key = lane >> 1, fb = (lane & 1) * 32;
    const v8h* vr = (const v8h*)(V + base + (size_t)key * 512 + fb);
    v8h* dst = (v8h*)&sV[slot][key * 64 + fb];
    dst[0] = vr[0]; dst[1] = vr[1]; dst[2] = vr[2]; dst[3] = vr[3];
  }
  __syncthreads();
  // O = P @ V  with P zero-padded K=16 -> 32
  v16h a2;
#pragma unroll
  for (int i = 0; i < 8; ++i) {
    a2[i]     = g ? (_Float16)0.f : sP[slot][mr * 16 + i];
    a2[8 + i] = g ? (_Float16)0.f : sP[slot][mr * 16 + 8 + i];
  }
  v8f o[4] = {};
#pragma unroll
  for (int t = 0; t < 4; ++t) {
    v16h bm;
    int f = t * 16 + mr;
    if (g == 0) {
#pragma unroll
      for (int i = 0; i < 16; ++i) bm[i] = sV[slot][i * 64 + f];
    } else {
#pragma unroll
      for (int i = 0; i < 16; ++i) bm[i] = (_Float16)0.f;   // padded K half
    }
    o[t] = __builtin_amdgcn_wmma_f32_16x16x32_f16(false, a2, false, bm, (short)0, o[t], false, false);
  }
#pragma unroll
  for (int t = 0; t < 4; ++t)
#pragma unroll
    for (int r = 0; r < 8; ++r)
      O[base + (size_t)(r + 8 * g) * 512 + t * 16 + mr] = (_Float16)o[t][r];
}

// ---------- row LayerNorm (+residual, +pos, +ELU), one wave per row ----------
__global__ __launch_bounds__(256) void k_ln(const float* __restrict__ in32,
                                            const float* __restrict__ res32,
                                            const float* __restrict__ gam,
                                            const float* __restrict__ bet,
                                            const float* __restrict__ pos,
                                            int N,
                                            float* __restrict__ out32,
                                            _Float16* __restrict__ out16,
                                            int o16stride, int o16off, int act) {
  int row  = blockIdx.x * 8 + (threadIdx.x >> 5);
  int lane = threadIdx.x & 31;
  int per  = N >> 5;
  const float* ip = in32 + (size_t)row * N;
  const float* rp = res32 ? res32 + (size_t)row * N : nullptr;
  float vals[32];
  float s = 0.f;
  for (int j = 0; j < per; ++j) {
    int c = lane + (j << 5);
    float v = ip[c];
    if (rp) v += rp[c];
    vals[j] = v; s += v;
  }
#pragma unroll
  for (int m = 16; m >= 1; m >>= 1) s += __shfl_xor(s, m, 32);
  float mu = s / (float)N;
  float q = 0.f;
  for (int j = 0; j < per; ++j) { float d = vals[j] - mu; q += d * d; }
#pragma unroll
  for (int m = 16; m >= 1; m >>= 1) q += __shfl_xor(q, m, 32);
  float rstd = rsqrtf(q / (float)N + 1e-5f);
  for (int j = 0; j < per; ++j) {
    int c = lane + (j << 5);
    float v = (vals[j] - mu) * rstd * gam[c] + bet[c];
    if (pos) v += pos[(row & 15) * N + c];
    if (act == 2) v = v > 0.f ? v : (__expf(v) - 1.f);
    if (out32) out32[(size_t)row * N + c] = v;
    if (out16) out16[(size_t)row * o16stride + o16off + c] = (_Float16)v;
  }
}

// ---------- GCN adjacency aggregate: out[b,m,d] = sum_n adj[m,n]*in[b,n,d] + bias[d] ----------
__global__ __launch_bounds__(256) void k_agg(const float* __restrict__ in32,
                                             const float* __restrict__ adj,
                                             const float* __restrict__ bias,
                                             float* __restrict__ out32, int N) {
  __shared__ float sadj[256];
  sadj[threadIdx.x] = adj[threadIdx.x];
  __syncthreads();
  int i = blockIdx.x * 256 + threadIdx.x;  // over B*N
  int b = i / N, d = i % N;
  const float* ip = in32 + (size_t)b * 16 * N + d;
  float v[16];
#pragma unroll
  for (int n = 0; n < 16; ++n) v[n] = ip[(size_t)n * N];
  float bv = bias[d];
  float* op = out32 + (size_t)b * 16 * N + d;
#pragma unroll
  for (int m = 0; m < 16; ++m) {
    float a = bv;
#pragma unroll
    for (int n = 0; n < 16; ++n) a += sadj[m * 16 + n] * v[n];
    op[(size_t)m * N] = a;
  }
}

// ---------- InstanceNorm over 16 spatial positions + ELU -> d_out [B,512,4,4] ----------
__global__ __launch_bounds__(256) void k_inorm(const float* __restrict__ y,
                                               float* __restrict__ out) {
  int i = blockIdx.x * 256 + threadIdx.x;  // B*512
  int b = i >> 9, c = i & 511;
  const float* ip = y + ((size_t)b * 16) * 512 + c;
  float v[16], s = 0.f;
#pragma unroll
  for (int t = 0; t < 16; ++t) { v[t] = ip[(size_t)t * 512]; s += v[t]; }
  float mu = s * (1.f / 16.f);
  float q = 0.f;
#pragma unroll
  for (int t = 0; t < 16; ++t) { float d = v[t] - mu; q += d * d; }
  float rstd = rsqrtf(q * (1.f / 16.f) + 1e-5f);
  float* op = out + ((size_t)b * 512 + c) * 16;
#pragma unroll
  for (int t = 0; t < 16; ++t) {
    float z = (v[t] - mu) * rstd;
    op[t] = z > 0.f ? z : (__expf(z) - 1.f);
  }
}

extern "C" void kernel_launch(void* const* d_in, const int* in_sizes, int n_in,
                              void* d_out, int out_size, void* d_ws, size_t ws_size,
                              hipStream_t stream) {
  (void)in_sizes; (void)n_in; (void)out_size; (void)ws_size;
  const float* x        = (const float*)d_in[0];
  const float* emb_w    = (const float*)d_in[1];
  const float* emb_b    = (const float*)d_in[2];
  const float* emb_ln_g = (const float*)d_in[3];
  const float* emb_ln_b = (const float*)d_in[4];
  const float* pos      = (const float*)d_in[5];
  const float* attn_wq  = (const float*)d_in[6];
  const float* attn_wk  = (const float*)d_in[7];
  const float* attn_wv  = (const float*)d_in[8];
  const float* attn_bq  = (const float*)d_in[9];
  const float* attn_bk  = (const float*)d_in[10];
  const float* attn_bv  = (const float*)d_in[11];
  const float* attn_wo  = (const float*)d_in[12];
  const float* attn_bo  = (const float*)d_in[13];
  const float* ln1_g    = (const float*)d_in[14];
  const float* ln1_b    = (const float*)d_in[15];
  const float* ff_w1    = (const float*)d_in[16];
  const float* ff_b1    = (const float*)d_in[17];
  const float* ff_w2    = (const float*)d_in[18];
  const float* ff_b2    = (const float*)d_in[19];
  const float* ln2_g    = (const float*)d_in[20];
  const float* ln2_b    = (const float*)d_in[21];
  const float* fc_w     = (const float*)d_in[22];
  const float* fc_b     = (const float*)d_in[23];
  const float* fc_ln_g  = (const float*)d_in[24];
  const float* fc_ln_b  = (const float*)d_in[25];
  const float* gcn_w1   = (const float*)d_in[26];
  const float* gcn_b1   = (const float*)d_in[27];
  const float* gcn_ln1g = (const float*)d_in[28];
  const float* gcn_ln1b = (const float*)d_in[29];
  const float* gcn_w2   = (const float*)d_in[30];
  const float* gcn_b2   = (const float*)d_in[31];
  const float* gcn_ln2g = (const float*)d_in[32];
  const float* gcn_ln2b = (const float*)d_in[33];
  const float* fusion_w = (const float*)d_in[34];
  const float* fusion_b = (const float*)d_in[35];
  const float* adj      = (const float*)d_in[36];

  char* ws = (char*)d_ws;
  size_t off = 0;
  auto alloc = [&](size_t bytes) { size_t r = off; off += (bytes + 255) & ~(size_t)255; return r; };

  // f16 weight arena
  _Float16* wbase = (_Float16*)(ws + alloc(10485760ULL * 2));
  size_t wo = 0;
  _Float16* w_emb = wbase + wo; wo += 262144;
  _Float16 *w_q[NLAYER], *w_k[NLAYER], *w_v[NLAYER], *w_o[NLAYER], *w_f1[NLAYER], *w_f2[NLAYER];
  for (int i = 0; i < NLAYER; ++i) {
    w_q[i]  = wbase + wo; wo += 262144;
    w_k[i]  = wbase + wo; wo += 262144;
    w_v[i]  = wbase + wo; wo += 262144;
    w_o[i]  = wbase + wo; wo += 262144;
    w_f1[i] = wbase + wo; wo += 524288;
    w_f2[i] = wbase + wo; wo += 524288;
  }
  _Float16* w_fc = wbase + wo; wo += 262144;
  _Float16* w_g1 = wbase + wo; wo += 524288;
  _Float16* w_g2 = wbase + wo; wo += 524288;
  _Float16* w_fu = wbase + wo; wo += 524288;

  _Float16* xtok16 = (_Float16*)(ws + alloc((size_t)BT * 512 * 2));
  _Float16* h16    = (_Float16*)(ws + alloc((size_t)BT * 512 * 2));
  float*    hf32   = (float*)   (ws + alloc((size_t)BT * 512 * 4));   // later reused as cat16
  _Float16* cat16  = (_Float16*)hf32;                                  // [BT,1024] f16, same bytes
  float*    tmp32  = (float*)   (ws + alloc((size_t)BT * 1024 * 4));
  _Float16* q16    = (_Float16*)(ws + alloc((size_t)BT * 512 * 2 * 4)); // q|k|v|o contiguous
  _Float16* k16 = q16 + (size_t)BT * 512;
  _Float16* v16 = k16 + (size_t)BT * 512;
  _Float16* o16 = v16 + (size_t)BT * 512;
  float*    tmp32b = (float*)q16;                                      // reused after attention
  _Float16* ffh16  = (_Float16*)(ws + alloc((size_t)BT * 1024 * 2));

  auto cvt = [&](const float* s, _Float16* d, int K, int N, int tr) {
    k_cvt<<<(K * N + 255) / 256, 256, 0, stream>>>(s, d, K, N, tr);
  };
  auto gemm = [&](const _Float16* A, const _Float16* W, const float* bias,
                  float* o32, _Float16* of16, int M, int K, int N, int act) {
    int blocks = (M / 128) * (N / 64);
    k_gemm<<<blocks, 128, 0, stream>>>(A, W, bias, o32, of16, M, K, N, act);
  };
  auto ln = [&](const float* in32, const float* res32, const float* g, const float* b,
                const float* p, int N, float* o32, _Float16* of16,
                int stride, int coloff, int act) {
    k_ln<<<BT / 8, 256, 0, stream>>>(in32, res32, g, b, p, N, o32, of16, stride, coloff, act);
  };

  // ---- weight conversion (all small) ----
  cvt(emb_w, w_emb, 512, 512, 1);
  for (int i = 0; i < NLAYER; ++i) {
    cvt(attn_wq + (size_t)i * 262144, w_q[i], 512, 512, 1);
    cvt(attn_wk + (size_t)i * 262144, w_k[i], 512, 512, 1);
    cvt(attn_wv + (size_t)i * 262144, w_v[i], 512, 512, 1);
    cvt(attn_wo + (size_t)i * 262144, w_o[i], 512, 512, 1);
    cvt(ff_w1   + (size_t)i * 524288, w_f1[i], 512, 1024, 1);
    cvt(ff_w2   + (size_t)i * 524288, w_f2[i], 1024, 512, 1);
  }
  cvt(fc_w, w_fc, 512, 512, 1);
  cvt(gcn_w1, w_g1, 512, 1024, 1);
  cvt(gcn_w2, w_g2, 1024, 512, 1);
  cvt(fusion_w, w_fu, 512, 1024, 0);  // already [N=512, K=1024]

  // ---- tokens ----
  k_pack<<<(BATCH * 512 * 16) / 256, 256, 0, stream>>>(x, xtok16);

  // ---- transformer: embedding ----
  gemm(xtok16, w_emb, emb_b, tmp32, nullptr, BT, 512, 512, 0);
  ln(tmp32, nullptr, emb_ln_g, emb_ln_b, pos, 512, hf32, h16, 512, 0, 0);

  // ---- transformer layers ----
  for (int i = 0; i < NLAYER; ++i) {
    gemm(h16, w_q[i], attn_bq + i * 512, nullptr, q16, BT, 512, 512, 0);
    gemm(h16, w_k[i], attn_bk + i * 512, nullptr, k16, BT, 512, 512, 0);
    gemm(h16, w_v[i], attn_bv + i * 512, nullptr, v16, BT, 512, 512, 0);
    k_attn<<<BATCH * NHEAD / 8, 256, 0, stream>>>(q16, k16, v16, o16);
    gemm(o16, w_o[i], attn_bo + i * 512, tmp32, nullptr, BT, 512, 512, 0);
    ln(tmp32, hf32, ln1_g + i * 512, ln1_b + i * 512, nullptr, 512, hf32, h16, 512, 0, 0);
    gemm(h16, w_f1[i], ff_b1 + i * 1024, nullptr, ffh16, BT, 512, 1024, 1);   // ReLU fused
    gemm(ffh16, w_f2[i], ff_b2 + i * 512, tmp32, nullptr, BT, 1024, 512, 0);
    ln(tmp32, hf32, ln2_g + i * 512, ln2_b + i * 512, nullptr, 512, hf32, h16, 512, 0, 0);
  }

  // ---- fc head -> cat16[:, :512] ----
  gemm(h16, w_fc, fc_b, tmp32, nullptr, BT, 512, 512, 0);
  ln(tmp32, nullptr, fc_ln_g, fc_ln_b, nullptr, 512, nullptr, cat16, 1024, 0, 0);

  // ---- GCN branch (q/k/v/o region reused as f32 aggregate buffer) ----
  gemm(xtok16, w_g1, nullptr, tmp32, nullptr, BT, 512, 1024, 0);
  k_agg<<<(BATCH * 1024) / 256, 256, 0, stream>>>(tmp32, adj, gcn_b1, tmp32b, 1024);
  ln(tmp32b, nullptr, gcn_ln1g, gcn_ln1b, nullptr, 1024, nullptr, ffh16, 1024, 0, 2); // ELU
  gemm(ffh16, w_g2, nullptr, tmp32, nullptr, BT, 1024, 512, 0);
  k_agg<<<(BATCH * 512) / 256, 256, 0, stream>>>(tmp32, adj, gcn_b2, tmp32b, 512);
  ln(tmp32b, nullptr, gcn_ln2g, gcn_ln2b, nullptr, 512, nullptr, cat16, 1024, 512, 2); // ELU

  // ---- fusion 1x1 conv + InstanceNorm + ELU ----
  gemm(cat16, w_fu, fusion_b, tmp32, nullptr, BT, 1024, 512, 0);
  k_inorm<<<(BATCH * 512) / 256, 256, 0, stream>>>(tmp32, (float*)d_out);
}
